// AttentionTorch_84507776516221
// MI455X (gfx1250) — matrix operand order repaired and checked by reading, not compiled, not run on hardware
//
#include <hip/hip_runtime.h>
#include <hip/hip_bf16.h>

// ---------------------------------------------------------------------------
// Attention block for MI455X (gfx1250, wave32, WMMA).
//   x:[2,2048,1024] f32 -> LN -> Q/K/V proj (bf16 WMMA) -> flash attention
//   (bf16 WMMA, online softmax, 64 keys per update) -> fused (wo@w_out) GEMM
//   + bias -> f32 out.
// All B operands are laid out with the K-dimension contiguous so every WMMA
// fragment load is two aligned 16B vector loads (global_load_b128).
// ---------------------------------------------------------------------------

typedef __bf16 bf16_t;
typedef __attribute__((ext_vector_type(16))) __bf16 v16bf;
typedef __attribute__((ext_vector_type(8)))  __bf16 v8bf;
typedef __attribute__((ext_vector_type(8)))  float  v8f;

#define DIM      1024
#define N_SEQ    2048
#define BATCH    2
#define HEADS    16
#define DH       64
#define ROWS     (BATCH * N_SEQ)        // 4096
#define ATTN_SCALE 0.125f               // 64^-0.5
#define LN_EPS   1e-5f

#define WMMA_BF16(a, b, c) \
  __builtin_amdgcn_wmma_f32_16x16x32_bf16(false, (a), false, (b), (short)0, (c), false, false)

// ---------------------------------------------------------------------------
// Fragment loaders (documented CDNA5 16-bit WMMA layouts, wave32).
// A 16x32 (row-major, K contiguous): lane<16 -> row=lane&15,
// halves 0..7 = K k0+8h.., halves 8..15 = K k0+16+8h.. (h = lane>>4).
// ---------------------------------------------------------------------------
__device__ inline v16bf load_a_frag(const bf16_t* A, size_t row0, int lda, int k0) {
  int lane = threadIdx.x & 31;
  size_t m = row0 + (size_t)(lane & 15);
  int kb = k0 + 8 * (lane >> 4);
  const bf16_t* p = A + m * (size_t)lda + kb;
  v8bf lo = *(const v8bf*)p;
  v8bf hi = *(const v8bf*)(p + 16);
  v16bf f;
#pragma unroll
  for (int i = 0; i < 8; ++i) { f[i] = lo[i]; f[8 + i] = hi[i]; }
  return f;
}

// B 32x16: halves 0..7 = K rows 8h.., halves 8..15 = K rows 16+8h.. (h = lane>>4)
// (matches the A layout); column n = lane&15.  Source must have K contiguous:
//   B[kk][n] = Src[(n0 + n) * ld + (c0 + kk)]   -> two aligned 16B loads.
__device__ inline v16bf load_b_contigK(const bf16_t* Src, size_t n0, int ld, int c0) {
  int lane = threadIdx.x & 31;
  size_t n = n0 + (size_t)(lane & 15);
  int kb = c0 + 8 * (lane >> 4);
  const bf16_t* p = Src + n * (size_t)ld + kb;
  v8bf lo = *(const v8bf*)p;
  v8bf hi = *(const v8bf*)(p + 16);
  v16bf f;
#pragma unroll
  for (int i = 0; i < 8; ++i) { f[i] = lo[i]; f[8 + i] = hi[i]; }
  return f;
}

__device__ inline float half_max(float v) {   // reduce across 16 lanes of a half-wave
#pragma unroll
  for (int m = 1; m <= 8; m <<= 1) v = fmaxf(v, __shfl_xor(v, m, 32));
  return v;
}
__device__ inline float half_sum(float v) {
#pragma unroll
  for (int m = 1; m <= 8; m <<= 1) v += __shfl_xor(v, m, 32);
  return v;
}

// ---------------------------------------------------------------------------
// LayerNorm (f32 in) -> bf16 out.  One 256-thread block per row.
// ---------------------------------------------------------------------------
__global__ __launch_bounds__(256) void ln_cast_kernel(
    const float* __restrict__ x, const float* __restrict__ gamma,
    const float* __restrict__ beta, bf16_t* __restrict__ xn) {
  int row = blockIdx.x;
  const float* xr = x + (size_t)row * DIM;
  float s = 0.f, s2 = 0.f;
  for (int i = threadIdx.x; i < DIM; i += 256) {
    float v = xr[i];
    s += v; s2 += v * v;
  }
#pragma unroll
  for (int m = 1; m <= 16; m <<= 1) { s += __shfl_xor(s, m, 32); s2 += __shfl_xor(s2, m, 32); }
  __shared__ float sb[8], sb2[8];
  int wid = threadIdx.x >> 5;
  if ((threadIdx.x & 31) == 0) { sb[wid] = s; sb2[wid] = s2; }
  __syncthreads();
  float tot = 0.f, tot2 = 0.f;
#pragma unroll
  for (int i = 0; i < 8; ++i) { tot += sb[i]; tot2 += sb2[i]; }
  float mu = tot * (1.f / DIM);
  float var = tot2 * (1.f / DIM) - mu * mu;
  float rstd = rsqrtf(var + LN_EPS);
  bf16_t* xo = xn + (size_t)row * DIM;
  for (int i = threadIdx.x; i < DIM; i += 256)
    xo[i] = (bf16_t)((xr[i] - mu) * rstd * gamma[i] + beta[i]);
}

// Plain cast (row-major kept): used for the A-side weight wo.
__global__ __launch_bounds__(256) void cast_f32_bf16(
    const float* __restrict__ src, bf16_t* __restrict__ dst, int n) {
  for (int i = blockIdx.x * 256 + threadIdx.x; i < n; i += gridDim.x * 256)
    dst[i] = (bf16_t)src[i];
}

// Cast + transpose: src[Kd][Nd] f32 -> dst[Nd][Kd] bf16 (K made contiguous).
// LDS-tiled 32x32 so both global streams stay coalesced.
__global__ __launch_bounds__(256) void cast_transpose_f32_bf16(
    const float* __restrict__ src, bf16_t* __restrict__ dst, int Kd, int Nd) {
  __shared__ bf16_t tile[32][33];
  int n0 = blockIdx.x * 32, k0 = blockIdx.y * 32;
#pragma unroll
  for (int r = 0; r < 4; ++r) {
    int k = k0 + threadIdx.y + 8 * r;
    tile[threadIdx.y + 8 * r][threadIdx.x] = (bf16_t)src[(size_t)k * Nd + n0 + threadIdx.x];
  }
  __syncthreads();
#pragma unroll
  for (int r = 0; r < 4; ++r) {
    int n = n0 + threadIdx.y + 8 * r;
    dst[(size_t)n * Kd + k0 + threadIdx.x] = tile[threadIdx.x][threadIdx.y + 8 * r];
  }
}

// ---------------------------------------------------------------------------
// GEMM: C[M,N] = A[M,K](bf16, row-major) x Bt[N,K](bf16, K contiguous).
// One wave per 32x64 C-tile: 2 A-frags x 4 B-frags -> 8 WMMAs per K-step.
// OUT_F32: f32 + bias output.  STORE_T: write C transposed (Ct[N][M]).
// ---------------------------------------------------------------------------
template <bool OUT_F32, bool STORE_T>
__global__ __launch_bounds__(256) void gemm_wmma_bf16(
    const bf16_t* __restrict__ A, const bf16_t* __restrict__ Bt,
    void* __restrict__ Cout, const float* __restrict__ bias,
    int M, int N, int K) {
  int wid = threadIdx.x >> 5;
  int gw = blockIdx.x * (blockDim.x >> 5) + wid;
  int tilesN = N >> 6;
  int tm = gw / tilesN;
  int tn = gw - tm * tilesN;
  if (tm * 32 >= M) return;

  v8f acc[2][4] = {};
  for (int k0 = 0; k0 < K; k0 += 32) {
    v16bf a0 = load_a_frag(A, (size_t)tm * 32, K, k0);
    v16bf a1 = load_a_frag(A, (size_t)tm * 32 + 16, K, k0);
#pragma unroll
    for (int f = 0; f < 4; ++f) {
      v16bf b = load_b_contigK(Bt, (size_t)tn * 64 + f * 16, K, k0);
      acc[0][f] = WMMA_BF16(a0, b, acc[0][f]);
      acc[1][f] = WMMA_BF16(a1, b, acc[1][f]);
    }
  }
  int lane = threadIdx.x & 31;
  int nc = lane & 15, hh = lane >> 4;
#pragma unroll
  for (int r = 0; r < 2; ++r) {
#pragma unroll
    for (int f = 0; f < 4; ++f) {
#pragma unroll
      for (int j = 0; j < 8; ++j) {
        size_t row = (size_t)tm * 32 + r * 16 + j + 8 * hh;
        size_t col = (size_t)tn * 64 + f * 16 + nc;
        float v = acc[r][f][j];
        if constexpr (OUT_F32)      ((float*)Cout)[row * N + col] = v + bias[col];
        else if constexpr (STORE_T) ((bf16_t*)Cout)[col * M + row] = (bf16_t)v;
        else                        ((bf16_t*)Cout)[row * N + col] = (bf16_t)v;
      }
    }
  }
}

// ---------------------------------------------------------------------------
// Flash attention: one wave per (batch, head, 16-row q-tile); 4 waves/block.
// 64 keys per online-softmax update: 8 score WMMAs, one softmax pass
// (cross-lane shfl reductions in the wave32 C-layout), P staged through LDS
// (C-layout -> A-layout, s_wait_dscnt), 8 PV WMMAs.  V is pre-transposed
// (Vt[dh][row]) so its fragments load as two 16B vectors.
// ---------------------------------------------------------------------------
__global__ __launch_bounds__(128) void attn_flash_kernel(
    const bf16_t* __restrict__ Q, const bf16_t* __restrict__ Km,
    const bf16_t* __restrict__ Vt, bf16_t* __restrict__ Oout) {
  __shared__ __attribute__((aligned(16))) bf16_t Plds[4][16 * 64];
  int wid = threadIdx.x >> 5;
  int lane = threadIdx.x & 31;
  int tile = blockIdx.x * 4 + wid;           // 0 .. 4095
  int b   = tile >> 11;                      // 2048 tiles per batch
  int rem = tile & 2047;
  int h   = rem >> 7;                        // 128 q-tiles per head
  int qt  = rem & 127;
  size_t arow0 = (size_t)b * N_SEQ + (size_t)qt * 16;
  size_t brow0 = (size_t)b * N_SEQ;
  int fbase = h * DH;

  v16bf aq0 = load_a_frag(Q, arow0, DIM, fbase);
  v16bf aq1 = load_a_frag(Q, arow0, DIM, fbase + 32);

  float mrun[8], lrun[8];
#pragma unroll
  for (int j = 0; j < 8; ++j) { mrun[j] = -INFINITY; lrun[j] = 0.f; }
  v8f o[4] = {};

  for (int kb = 0; kb < N_SEQ; kb += 64) {
    // ---- scores: S[16 q x 64 keys]; K rows have dh contiguous ----
    v8f c[4] = {};
#pragma unroll
    for (int g = 0; g < 4; ++g) {
      v16bf bk = load_b_contigK(Km, brow0 + kb + g * 16, DIM, fbase);
      c[g] = WMMA_BF16(aq0, bk, c[g]);
      bk = load_b_contigK(Km, brow0 + kb + g * 16, DIM, fbase + 32);
      c[g] = WMMA_BF16(aq1, bk, c[g]);
    }

    // ---- online softmax; row j+8*(lane>>4) lives in VGPR j across a half ----
    float corr[8];
#pragma unroll
    for (int j = 0; j < 8; ++j) {
      float s0 = c[0][j] * ATTN_SCALE;
      float s1 = c[1][j] * ATTN_SCALE;
      float s2 = c[2][j] * ATTN_SCALE;
      float s3 = c[3][j] * ATTN_SCALE;
      float rmax = half_max(fmaxf(fmaxf(s0, s1), fmaxf(s2, s3)));
      float mnew = fmaxf(mrun[j], rmax);
      corr[j] = __expf(mrun[j] - mnew);
      float p0 = __expf(s0 - mnew);
      float p1 = __expf(s1 - mnew);
      float p2 = __expf(s2 - mnew);
      float p3 = __expf(s3 - mnew);
      lrun[j] = lrun[j] * corr[j] + half_sum((p0 + p1) + (p2 + p3));
      mrun[j] = mnew;
      int m = j + 8 * (lane >> 4);
      bf16_t* prow = &Plds[wid][m * 64 + (lane & 15)];
      prow[0]  = (bf16_t)p0;
      prow[16] = (bf16_t)p1;
      prow[32] = (bf16_t)p2;
      prow[48] = (bf16_t)p3;
    }
#pragma unroll
    for (int t = 0; t < 4; ++t)
#pragma unroll
      for (int j = 0; j < 8; ++j) o[t][j] *= corr[j];

    // make all lanes' P stores visible before the A-layout reload
    asm volatile("s_wait_dscnt 0" ::: "memory");
    v16bf pa0 = load_a_frag(&Plds[wid][0], 0, 64, 0);
    v16bf pa1 = load_a_frag(&Plds[wid][0], 0, 64, 32);

    // ---- O += P (16x64 keys) x V (64 keys x 64 dh); Vt has keys contiguous ----
#pragma unroll
    for (int t = 0; t < 4; ++t) {
      v16bf vb = load_b_contigK(Vt, (size_t)(fbase + t * 16), ROWS, (int)(brow0 + kb));
      o[t] = WMMA_BF16(pa0, vb, o[t]);
      vb = load_b_contigK(Vt, (size_t)(fbase + t * 16), ROWS, (int)(brow0 + kb) + 32);
      o[t] = WMMA_BF16(pa1, vb, o[t]);
    }
  }

  int nc = lane & 15, hh = lane >> 4;
#pragma unroll
  for (int j = 0; j < 8; ++j) {
    float inv = 1.f / lrun[j];
#pragma unroll
    for (int t = 0; t < 4; ++t) {
      size_t row = arow0 + j + 8 * hh;
      size_t col = (size_t)fbase + t * 16 + nc;
      Oout[row * DIM + col] = (bf16_t)(o[t][j] * inv);
    }
  }
}

// ---------------------------------------------------------------------------
// Host launcher.  Workspace layout (~52 MB, all offsets 16B-aligned):
//   xn(8M) | woA(2M) | wqT,wkT,wvT,woutT (4x2M) | WcT(2M) | Q(8M) | K(8M)
//   | Vt(8M) | attn(8M)
// ---------------------------------------------------------------------------
extern "C" void kernel_launch(void* const* d_in, const int* in_sizes, int n_in,
                              void* d_out, int out_size, void* d_ws, size_t ws_size,
                              hipStream_t stream) {
  const float* x     = (const float*)d_in[0];
  const float* gamma = (const float*)d_in[1];
  const float* beta  = (const float*)d_in[2];
  const float* wq    = (const float*)d_in[3];
  const float* wk    = (const float*)d_in[4];
  const float* wv    = (const float*)d_in[5];
  const float* wo    = (const float*)d_in[6];
  const float* wout  = (const float*)d_in[7];
  const float* bout  = (const float*)d_in[8];

  const size_t ACT = (size_t)ROWS * DIM * sizeof(bf16_t);  // 8 MB
  const size_t WGT = (size_t)DIM * DIM * sizeof(bf16_t);   // 2 MB
  char* ws = (char*)d_ws;
  size_t off = 0;
  bf16_t* xn     = (bf16_t*)(ws + off); off += ACT;
  bf16_t* wo_a   = (bf16_t*)(ws + off); off += WGT;   // row-major (A operand)
  bf16_t* wq_t   = (bf16_t*)(ws + off); off += WGT;   // transposed (B operand)
  bf16_t* wk_t   = (bf16_t*)(ws + off); off += WGT;
  bf16_t* wv_t   = (bf16_t*)(ws + off); off += WGT;
  bf16_t* wout_t = (bf16_t*)(ws + off); off += WGT;
  bf16_t* wc_t   = (bf16_t*)(ws + off); off += WGT;   // (wo@w_out)^T
  bf16_t* Qb     = (bf16_t*)(ws + off); off += ACT;   // row-major
  bf16_t* Kb     = (bf16_t*)(ws + off); off += ACT;   // row-major (dh contiguous)
  bf16_t* Vtb    = (bf16_t*)(ws + off); off += ACT;   // transposed (keys contiguous)
  bf16_t* Ab     = (bf16_t*)(ws + off); off += ACT;
  (void)ws_size; (void)in_sizes; (void)n_in; (void)out_size;

  dim3 tb(32, 8), tg(DIM / 32, DIM / 32);

  // 1) LayerNorm + cast
  ln_cast_kernel<<<ROWS, 256, 0, stream>>>(x, gamma, beta, xn);

  // 2) weight casts: wo stays row-major (A side); others transposed (B side)
  cast_f32_bf16<<<512, 256, 0, stream>>>(wo, wo_a, DIM * DIM);
  cast_transpose_f32_bf16<<<tg, tb, 0, stream>>>(wq,   wq_t,   DIM, DIM);
  cast_transpose_f32_bf16<<<tg, tb, 0, stream>>>(wk,   wk_t,   DIM, DIM);
  cast_transpose_f32_bf16<<<tg, tb, 0, stream>>>(wv,   wv_t,   DIM, DIM);
  cast_transpose_f32_bf16<<<tg, tb, 0, stream>>>(wout, wout_t, DIM, DIM);

  // 3) fold tail projections: WcT = (wo @ w_out)^T   (stored K-contiguous)
  gemm_wmma_bf16<false, true><<<64, 256, 0, stream>>>(
      wo_a, wout_t, (void*)wc_t, nullptr, DIM, DIM, DIM);

  // 4) Q/K/V projections; V written transposed for the attention PV stage
  gemm_wmma_bf16<false, false><<<256, 256, 0, stream>>>(xn, wq_t, (void*)Qb,  nullptr, ROWS, DIM, DIM);
  gemm_wmma_bf16<false, false><<<256, 256, 0, stream>>>(xn, wk_t, (void*)Kb,  nullptr, ROWS, DIM, DIM);
  gemm_wmma_bf16<false, true ><<<256, 256, 0, stream>>>(xn, wv_t, (void*)Vtb, nullptr, ROWS, DIM, DIM);

  // 5) flash attention: 4096 q-tiles, 4 waves/block
  attn_flash_kernel<<<1024, 128, 0, stream>>>(Qb, Kb, Vtb, Ab);

  // 6) out = attn @ Wc + b_out  (f32 output)
  gemm_wmma_bf16<true, false><<<256, 256, 0, stream>>>(Ab, wc_t, d_out, bout, ROWS, DIM, DIM);
}
